// MHA_24386824307060
// MI455X (gfx1250) — matrix operand order repaired and checked by reading, not compiled, not run on hardware
//
#include <hip/hip_runtime.h>

#define NH   16
#define HD   64
#define INSZ 1024
#define NB   4
#define SEQ  2048
#define NROW (NB * SEQ)          // 8192
#define ATT_SCALE 0.125f         // 1/sqrt(64)

typedef __attribute__((ext_vector_type(16))) __bf16 v16bf;
typedef __attribute__((ext_vector_type(8)))  float  v8f;

struct alignas(16) BF8 { __bf16 v[8]; };
struct alignas(8)  BF4 { __bf16 v[4]; };

#define WMMA_KERNEL __launch_bounds__(256) __attribute__((amdgpu_waves_per_eu(2)))

// ---------------------------------------------------------------- WMMA helpers

__device__ __forceinline__ v8f wmma_bf16(v16bf a, v16bf b, v8f c) {
  // v_wmma_f32_16x16x32_bf16: D = A(16x32) * B(32x16) + C(16x16 f32)
  return __builtin_amdgcn_wmma_f32_16x16x32_bf16(false, a, false, b,
                                                 (short)0, c, false, false);
}

// A fragment 16x32 (MxK), bf16, from row-major [rows, ld] storage.
// lane l: row = row0 + (l&15); elems 0..7  -> K = k0 + (l>>4)*8 + i
//                              elems 8..15 -> K = k0 + 16 + (l>>4)*8 + i
__device__ __forceinline__ v16bf load_a_frag(const __bf16* __restrict__ base,
                                             int ld, int row0, int k0) {
  const int lane = threadIdx.x & 31;
  const __bf16* p = base + (size_t)(row0 + (lane & 15)) * ld + k0 + ((lane >> 4) << 3);
  BF8 lo = *(const BF8*)p;
  BF8 hi = *(const BF8*)(p + 16);
  v16bf a;
#pragma unroll
  for (int i = 0; i < 8; ++i) { a[i] = lo.v[i]; a[8 + i] = hi.v[i]; }
  return a;
}

// B fragment 32x16 (KxN), bf16, from storage where each matrix COLUMN n is a
// contiguous row of length ld ("[N,K]" layout).
// lane l: n = n0 + (l&15); elems 0..7  -> K = k0 + (l>>4)*8 + i
//                         elems 8..15 -> K = k0 + 16 + (l>>4)*8 + i
__device__ __forceinline__ v16bf load_b_frag(const __bf16* __restrict__ base,
                                             int ld, int n0, int k0) {
  const int lane = threadIdx.x & 31;
  const __bf16* p = base + (size_t)(n0 + (lane & 15)) * ld + k0 + ((lane >> 4) << 3);
  BF8 lo = *(const BF8*)p;
  BF8 hi = *(const BF8*)(p + 16);
  v16bf b;
#pragma unroll
  for (int i = 0; i < 8; ++i) { b[i] = lo.v[i]; b[8 + i] = hi.v[i]; }
  return b;
}

// reductions across the 16 lanes of a wave32 half (masks never cross bit 4)
__device__ __forceinline__ float half_max(float v) {
#pragma unroll
  for (int m = 1; m < 16; m <<= 1) v = fmaxf(v, __shfl_xor(v, m, 32));
  return v;
}
__device__ __forceinline__ float half_sum(float v) {
#pragma unroll
  for (int m = 1; m < 16; m <<= 1) v += __shfl_xor(v, m, 32);
  return v;
}

// ---------------------------------------------------------------- conversions

__global__ void k_cvt_x(const float* __restrict__ x, __bf16* __restrict__ xb) {
  const int t = blockIdx.x * blockDim.x + threadIdx.x;
  const float4 f = ((const float4*)x)[t];
  BF4 o;
  o.v[0] = (__bf16)f.x; o.v[1] = (__bf16)f.y;
  o.v[2] = (__bf16)f.z; o.v[3] = (__bf16)f.w;
  ((BF4*)xb)[t] = o;
}

// dst[n*1024 + k] = (bf16) src[k*1024 + n]   (LDS-tiled 32x32 transpose)
__global__ void k_cvt_wT(const float* __restrict__ src, __bf16* __restrict__ dst) {
  __shared__ float tile[32][33];
  const int bx = blockIdx.x & 31, by = blockIdx.x >> 5;
  const int tx = threadIdx.x & 31, ty = threadIdx.x >> 5;  // 32 x 8
#pragma unroll
  for (int i = ty; i < 32; i += 8)
    tile[i][tx] = src[(size_t)(by * 32 + i) * INSZ + bx * 32 + tx];
  __syncthreads();
#pragma unroll
  for (int i = ty; i < 32; i += 8)
    dst[(size_t)(bx * 32 + i) * INSZ + by * 32 + tx] = (__bf16)tile[tx][i];
}

// ---------------------------------------------------------------- QKV GEMM
// 16x64 strip per wave. A-frag double-buffered across K-steps; B-frags loaded
// just-in-time into 4 distinct locals so the loads clause together and the
// WMMAs take staggered partial waits.

__global__ void WMMA_KERNEL
k_qkv(const __bf16* __restrict__ xb,
      const __bf16* __restrict__ wqT, const __bf16* __restrict__ wkT,
      const __bf16* __restrict__ wvT,
      const float* __restrict__ bq, const float* __restrict__ bk,
      const float* __restrict__ bv,
      __bf16* __restrict__ qo, __bf16* __restrict__ ko,
      __bf16* __restrict__ vT) {
  const int lane = threadIdx.x & 31;
  const int t = blockIdx.x * 8 + (threadIdx.x >> 5);
  const int tm = t / 48, tg = t % 48;         // 512 x 48 strips over [8192, 3072]
  const int m0 = tm * 16;
  const int n0g = tg * 64;
  const int sel = n0g >> 10;                  // 0:Q 1:K 2:V (uniform per strip)
  const int n0 = n0g & 1023;                  // multiple of 64 -> one head
  const __bf16* wT = (sel == 0) ? wqT : (sel == 1) ? wkT : wvT;
  const float* bias = (sel == 0) ? bq : (sel == 1) ? bk : bv;

  v8f a0 = {}, a1 = {}, a2 = {}, a3 = {};
  v16bf af = load_a_frag(xb, INSZ, m0, 0);
  for (int kk = 0; kk < INSZ - 32; kk += 32) {
    const v16bf c0 = load_b_frag(wT, INSZ, n0,      kk);
    const v16bf c1 = load_b_frag(wT, INSZ, n0 + 16, kk);
    const v16bf c2 = load_b_frag(wT, INSZ, n0 + 32, kk);
    const v16bf c3 = load_b_frag(wT, INSZ, n0 + 48, kk);
    const v16bf an = load_a_frag(xb, INSZ, m0, kk + 32);
    a0 = wmma_bf16(af, c0, a0);
    a1 = wmma_bf16(af, c1, a1);
    a2 = wmma_bf16(af, c2, a2);
    a3 = wmma_bf16(af, c3, a3);
    af = an;
  }
  {
    const v16bf c0 = load_b_frag(wT, INSZ, n0,      INSZ - 32);
    const v16bf c1 = load_b_frag(wT, INSZ, n0 + 16, INSZ - 32);
    const v16bf c2 = load_b_frag(wT, INSZ, n0 + 32, INSZ - 32);
    const v16bf c3 = load_b_frag(wT, INSZ, n0 + 48, INSZ - 32);
    a0 = wmma_bf16(af, c0, a0);
    a1 = wmma_bf16(af, c1, a1);
    a2 = wmma_bf16(af, c2, a2);
    a3 = wmma_bf16(af, c3, a3);
  }

  const int c = lane & 15;
  const int h = n0 >> 6;
  const int rbase = m0 + ((lane >> 4) << 3);
  const int bidx = rbase >> 11;               // whole strip within one batch
  const int s0 = rbase & 2047;
  const float b0 = bias[n0 + c],      b1 = bias[n0 + 16 + c];
  const float b2 = bias[n0 + 32 + c], b3 = bias[n0 + 48 + c];

  if (sel < 2) {
    __bf16* dst = ((sel == 0) ? qo : ko) + (size_t)(bidx * NH + h) * SEQ * HD;
#pragma unroll
    for (int r = 0; r < 8; ++r) {
      __bf16* p = dst + (size_t)(s0 + r) * HD;
      p[c]      = (__bf16)(a0[r] + b0);
      p[16 + c] = (__bf16)(a1[r] + b1);
      p[32 + c] = (__bf16)(a2[r] + b2);
      p[48 + c] = (__bf16)(a3[r] + b3);
    }
  } else {
    __bf16* vb = vT + (size_t)(bidx * NH + h) * HD * SEQ;   // [d][s], contiguous in s
    BF8 o0, o1, o2, o3;
#pragma unroll
    for (int r = 0; r < 8; ++r) {
      o0.v[r] = (__bf16)(a0[r] + b0);
      o1.v[r] = (__bf16)(a1[r] + b1);
      o2.v[r] = (__bf16)(a2[r] + b2);
      o3.v[r] = (__bf16)(a3[r] + b3);
    }
    *(BF8*)(vb + (size_t)(c)      * SEQ + s0) = o0;
    *(BF8*)(vb + (size_t)(16 + c) * SEQ + s0) = o1;
    *(BF8*)(vb + (size_t)(32 + c) * SEQ + s0) = o2;
    *(BF8*)(vb + (size_t)(48 + c) * SEQ + s0) = o3;
  }
}

// ---------------------------------------------------------------- flash attention
// K and V fragments for the current key block are issued together at the loop
// top; V latency is hidden by the softmax VALU block + LDS round trip, K-load
// latency overlaps the previous iteration's P@V WMMAs.

__global__ void WMMA_KERNEL
k_flash(const __bf16* __restrict__ qb, const __bf16* __restrict__ kb,
        const __bf16* __restrict__ vT,
        float* __restrict__ mstat, float* __restrict__ lstat,
        __bf16* __restrict__ attnb) {
  __shared__ __bf16 plds[8][16 * 32];         // per-wave P tile (1 KB each)
  const int lane = threadIdx.x & 31;
  const int wv = threadIdx.x >> 5;
  const int w = blockIdx.x * 8 + wv;
  const int qt = w & 127;                     // 128 q-tiles of 16 rows
  const int bh = w >> 7;
  const int bat = bh >> 4, h = bh & 15;
  const int qbase = qt * 16;
  const __bf16* qh = qb + (size_t)bh * SEQ * HD;
  const __bf16* kh = kb + (size_t)bh * SEQ * HD;
  const __bf16* vh = vT + (size_t)bh * HD * SEQ;

  const v16bf qa0 = load_a_frag(qh, HD, qbase, 0);
  const v16bf qa1 = load_a_frag(qh, HD, qbase, 32);

  float mr[8], lr[8];
#pragma unroll
  for (int r = 0; r < 8; ++r) { mr[r] = -1e30f; lr[r] = 0.f; }
  v8f o0 = {}, o1 = {}, o2 = {}, o3 = {};
  const int rh = (lane >> 4) << 3;            // C-frag row sub-base
  const int c = lane & 15;                    // C-frag column

  for (int kt = 0; kt < SEQ; kt += 32) {
    const v16bf k00 = load_b_frag(kh, HD, kt,      0);
    const v16bf k01 = load_b_frag(kh, HD, kt,      32);
    const v16bf k10 = load_b_frag(kh, HD, kt + 16, 0);
    const v16bf k11 = load_b_frag(kh, HD, kt + 16, 32);
    const v16bf vb0 = load_b_frag(vh, SEQ, 0,  kt);
    const v16bf vb1 = load_b_frag(vh, SEQ, 16, kt);
    const v16bf vb2 = load_b_frag(vh, SEQ, 32, kt);
    const v16bf vb3 = load_b_frag(vh, SEQ, 48, kt);

    v8f z = {};
    v8f s0 = wmma_bf16(qa0, k00, z); s0 = wmma_bf16(qa1, k01, s0);
    v8f s1 = wmma_bf16(qa0, k10, z); s1 = wmma_bf16(qa1, k11, s1);

    float p0[8], p1[8], alpha[8];
#pragma unroll
    for (int r = 0; r < 8; ++r) {
      float tmax = half_max(fmaxf(s0[r], s1[r]) * ATT_SCALE);
      const float mn = fmaxf(mr[r], tmax);
      alpha[r] = __expf(mr[r] - mn);
      p0[r] = __expf(s0[r] * ATT_SCALE - mn);
      p1[r] = __expf(s1[r] * ATT_SCALE - mn);
      lr[r] = lr[r] * alpha[r] + half_sum(p0[r] + p1[r]);
      mr[r] = mn;
    }
#pragma unroll
    for (int r = 0; r < 8; ++r) {
      o0[r] *= alpha[r]; o1[r] *= alpha[r]; o2[r] *= alpha[r]; o3[r] *= alpha[r];
    }
    // C-frag -> row-major LDS [16][32] -> A-frag (same wave: HW DScnt ordering)
    __bf16* pl = plds[wv];
#pragma unroll
    for (int r = 0; r < 8; ++r) {
      pl[(rh + r) * 32 + c]      = (__bf16)p0[r];
      pl[(rh + r) * 32 + 16 + c] = (__bf16)p1[r];
    }
    const int arow = lane & 15;
    const int ak = (lane >> 4) << 3;
    BF8 lo = *(const BF8*)&pl[arow * 32 + ak];
    BF8 hi = *(const BF8*)&pl[arow * 32 + 16 + ak];
    v16bf pa;
#pragma unroll
    for (int i = 0; i < 8; ++i) { pa[i] = lo.v[i]; pa[8 + i] = hi.v[i]; }

    o0 = wmma_bf16(pa, vb0, o0);
    o1 = wmma_bf16(pa, vb1, o1);
    o2 = wmma_bf16(pa, vb2, o2);
    o3 = wmma_bf16(pa, vb3, o3);
  }

  float rcp[8];
#pragma unroll
  for (int r = 0; r < 8; ++r) rcp[r] = 1.f / lr[r];
  if ((lane & 15) == 0) {                     // lanes 0 and 16 cover rows 0-7 / 8-15
    float* mp = mstat + (size_t)bh * SEQ + qbase + rh;
    float* lp = lstat + (size_t)bh * SEQ + qbase + rh;
#pragma unroll
    for (int r = 0; r < 8; ++r) { mp[r] = mr[r]; lp[r] = lr[r]; }
  }
  __bf16* ap = attnb + (size_t)(bat * SEQ + qbase + rh) * (NH * HD) + h * HD;
#pragma unroll
  for (int r = 0; r < 8; ++r) {
    const size_t ro = (size_t)r * (NH * HD);
    ap[ro + 0  + c] = (__bf16)(o0[r] * rcp[r]);
    ap[ro + 16 + c] = (__bf16)(o1[r] * rcp[r]);
    ap[ro + 32 + c] = (__bf16)(o2[r] * rcp[r]);
    ap[ro + 48 + c] = (__bf16)(o3[r] * rcp[r]);
  }
}

// ---------------------------------------------------------------- probs head-mean
// 16x64 strip per wave. Per head: all 8 key B-fragments are issued first as
// distinct values, then the 8 WMMAs, then the exp/accumulate VALU block.

__global__ void WMMA_KERNEL
k_probs(const __bf16* __restrict__ qb, const __bf16* __restrict__ kb,
        const float* __restrict__ mstat, const float* __restrict__ lstat,
        float* __restrict__ pm) {
  const int lane = threadIdx.x & 31;
  const int w = blockIdx.x * 8 + (threadIdx.x >> 5);
  const int kt = w & 31;                      // 32 key strips of 64
  const int qt = (w >> 5) & 127;              // 128 q-tiles of 16
  const int bat = w >> 12;
  const int qbase = qt * 16, kbase = kt * 64;
  const int rh = (lane >> 4) << 3;
  const int c = lane & 15;

  float acc[4][8] = {};
  for (int h = 0; h < NH; ++h) {
    const int bh = bat * NH + h;
    const __bf16* qh = qb + (size_t)bh * SEQ * HD;
    const __bf16* kh = kb + (size_t)bh * SEQ * HD;
    const v16bf qa0 = load_a_frag(qh, HD, qbase, 0);
    const v16bf qa1 = load_a_frag(qh, HD, qbase, 32);
    v16bf kb0[4], kb1[4];
#pragma unroll
    for (int n = 0; n < 4; ++n) {
      kb0[n] = load_b_frag(kh, HD, kbase + n * 16, 0);
      kb1[n] = load_b_frag(kh, HD, kbase + n * 16, 32);
    }
    float mh[8], rl[8];
    const float* mp = mstat + (size_t)bh * SEQ + qbase + rh;
    const float* lp = lstat + (size_t)bh * SEQ + qbase + rh;
#pragma unroll
    for (int r = 0; r < 8; ++r) { mh[r] = mp[r]; rl[r] = 1.f / lp[r]; }
    v8f s[4];
#pragma unroll
    for (int n = 0; n < 4; ++n) {
      v8f z = {};
      s[n] = wmma_bf16(qa0, kb0[n], z);
      s[n] = wmma_bf16(qa1, kb1[n], s[n]);
    }
#pragma unroll
    for (int n = 0; n < 4; ++n)
#pragma unroll
      for (int r = 0; r < 8; ++r)
        acc[n][r] += __expf(s[n][r] * ATT_SCALE - mh[r]) * rl[r];
  }
#pragma unroll
  for (int r = 0; r < 8; ++r) {
    float* prow = pm + (size_t)(bat * SEQ + qbase + rh + r) * SEQ + kbase;
#pragma unroll
    for (int n = 0; n < 4; ++n)
      prow[n * 16 + c] = acc[n][r] * (1.f / NH);
  }
}

// ---------------------------------------------------------------- output projection
// 16x64 strip per wave, same scheme as k_qkv.

__global__ void WMMA_KERNEL
k_oproj(const __bf16* __restrict__ attnb, const __bf16* __restrict__ woT,
        const float* __restrict__ bo, float* __restrict__ out) {
  const int lane = threadIdx.x & 31;
  const int t = blockIdx.x * 8 + (threadIdx.x >> 5);
  const int tm = t >> 4, tn = t & 15;         // 512 x 16 strips over [8192, 1024]
  const int m0 = tm * 16, n0 = tn * 64;
  v8f a0 = {}, a1 = {}, a2 = {}, a3 = {};
  v16bf af = load_a_frag(attnb, INSZ, m0, 0);
  for (int kk = 0; kk < INSZ - 32; kk += 32) {
    const v16bf c0 = load_b_frag(woT, INSZ, n0,      kk);
    const v16bf c1 = load_b_frag(woT, INSZ, n0 + 16, kk);
    const v16bf c2 = load_b_frag(woT, INSZ, n0 + 32, kk);
    const v16bf c3 = load_b_frag(woT, INSZ, n0 + 48, kk);
    const v16bf an = load_a_frag(attnb, INSZ, m0, kk + 32);
    a0 = wmma_bf16(af, c0, a0);
    a1 = wmma_bf16(af, c1, a1);
    a2 = wmma_bf16(af, c2, a2);
    a3 = wmma_bf16(af, c3, a3);
    af = an;
  }
  {
    const v16bf c0 = load_b_frag(woT, INSZ, n0,      INSZ - 32);
    const v16bf c1 = load_b_frag(woT, INSZ, n0 + 16, INSZ - 32);
    const v16bf c2 = load_b_frag(woT, INSZ, n0 + 32, INSZ - 32);
    const v16bf c3 = load_b_frag(woT, INSZ, n0 + 48, INSZ - 32);
    a0 = wmma_bf16(af, c0, a0);
    a1 = wmma_bf16(af, c1, a1);
    a2 = wmma_bf16(af, c2, a2);
    a3 = wmma_bf16(af, c3, a3);
  }

  const int c = lane & 15;
  const float b0 = bo[n0 + c],      b1 = bo[n0 + 16 + c];
  const float b2 = bo[n0 + 32 + c], b3 = bo[n0 + 48 + c];
  const int rbase = m0 + ((lane >> 4) << 3);
#pragma unroll
  for (int r = 0; r < 8; ++r) {
    float* p = out + (size_t)(rbase + r) * INSZ + n0;
    p[c]      = a0[r] + b0;
    p[16 + c] = a1[r] + b1;
    p[32 + c] = a2[r] + b2;
    p[48 + c] = a3[r] + b3;
  }
}

// ---------------------------------------------------------------- launcher

extern "C" void kernel_launch(void* const* d_in, const int* in_sizes, int n_in,
                              void* d_out, int out_size, void* d_ws, size_t ws_size,
                              hipStream_t stream) {
  (void)in_sizes; (void)n_in; (void)out_size; (void)ws_size;
  const float* x  = (const float*)d_in[0];
  const float* Wq = (const float*)d_in[1];
  const float* bq = (const float*)d_in[2];
  const float* Wk = (const float*)d_in[3];
  const float* bk = (const float*)d_in[4];
  const float* Wv = (const float*)d_in[5];
  const float* bv = (const float*)d_in[6];
  const float* Wo = (const float*)d_in[7];
  const float* bo = (const float*)d_in[8];

  float* out = (float*)d_out;                           // [4,2048,1024]
  float* pm  = out + (size_t)NROW * INSZ;               // [4,2048,2048]

  char* ws = (char*)d_ws;
  size_t off = 0;
  auto carve = [&](size_t bytes) {
    char* p = ws + off;
    off = (off + bytes + 255) & ~(size_t)255;
    return p;
  };
  __bf16* xb    = (__bf16*)carve((size_t)NROW * INSZ * 2);   // 16 MB
  __bf16* wqT   = (__bf16*)carve((size_t)INSZ * INSZ * 2);   // 2 MB
  __bf16* wkT   = (__bf16*)carve((size_t)INSZ * INSZ * 2);
  __bf16* wvT   = (__bf16*)carve((size_t)INSZ * INSZ * 2);
  __bf16* woT   = (__bf16*)carve((size_t)INSZ * INSZ * 2);
  __bf16* qbuf  = (__bf16*)carve((size_t)NB * NH * SEQ * HD * 2);  // 16 MB [b,h,s,d]
  __bf16* kbuf  = (__bf16*)carve((size_t)NB * NH * SEQ * HD * 2);  // 16 MB [b,h,s,d]
  __bf16* vTbuf = (__bf16*)carve((size_t)NB * NH * HD * SEQ * 2);  // 16 MB [b,h,d,s]
  __bf16* attnb = (__bf16*)carve((size_t)NROW * INSZ * 2);         // 16 MB
  float*  mstat = (float*)carve((size_t)NB * NH * SEQ * 4);
  float*  lstat = (float*)carve((size_t)NB * NH * SEQ * 4);

  // 1) precision conversion + weight transposes
  k_cvt_x<<<(NROW * INSZ) / (4 * 256), 256, 0, stream>>>(x, xb);
  k_cvt_wT<<<1024, 256, 0, stream>>>(Wq, wqT);
  k_cvt_wT<<<1024, 256, 0, stream>>>(Wk, wkT);
  k_cvt_wT<<<1024, 256, 0, stream>>>(Wv, wvT);
  k_cvt_wT<<<1024, 256, 0, stream>>>(Wo, woT);

  // 2) fused QKV projection (512*48 strips of 16x64, 8 waves/block)
  k_qkv<<<(512 * 48) / 8, 256, 0, stream>>>(xb, wqT, wkT, wvT, bq, bk, bv,
                                            qbuf, kbuf, vTbuf);

  // 3) flash attention (B*H*128 waves)
  k_flash<<<(NB * NH * (SEQ / 16)) / 8, 256, 0, stream>>>(qbuf, kbuf, vTbuf,
                                                          mstat, lstat, attnb);

  // 4) mean-over-heads probabilities (B*128*32 strips of 16x64)
  k_probs<<<(NB * 128 * 32) / 8, 256, 0, stream>>>(qbuf, kbuf, mstat, lstat, pm);

  // 5) output projection (512*16 strips of 16x64)
  k_oproj<<<(512 * 16) / 8, 256, 0, stream>>>(attnb, woT, bo, out);
}
